// CausalSelfAttention_44513041056202
// MI455X (gfx1250) — matrix operand order repaired and checked by reading, not compiled, not run on hardware
//
#include <hip/hip_runtime.h>
#include <hip/hip_bf16.h>

// ---------------------------------------------------------------------------
// CausalSelfAttention for MI455X (gfx1250), bf16 WMMA + flash-attention.
// B=2, T=2048, C=1024, H=16, D=64.  scale = 1/sqrt(C) = 1/32.
// ---------------------------------------------------------------------------

typedef __attribute__((ext_vector_type(16))) __bf16 v16bf;
typedef __attribute__((ext_vector_type(8)))  __bf16 v8bf;
typedef __attribute__((ext_vector_type(8)))  float  v8f;

union V16 { v16bf v; v8bf h[2]; };

// gfx1250 async LDS copy path (ASYNCcnt-tracked, no VGPR staging), if declared
#if defined(__has_builtin)
#  if __has_builtin(__builtin_amdgcn_global_load_async_to_lds_b128)
#    define USE_ASYNC_LDS 1
#  endif
#endif
#ifndef USE_ASYNC_LDS
#  define USE_ASYNC_LDS 0
#endif

#if USE_ASYNC_LDS
// exact param type per compiler diagnostic: 'int __vector(4) __device__ *'
typedef int v4i_ __attribute__((vector_size(16)));
typedef __attribute__((address_space(1))) v4i_ g1_v4i;
typedef __attribute__((address_space(3))) v4i_ l3_v4i;

__device__ __forceinline__ g1_v4i* as_global_v4i(const void* p) {
    // const-cast, then addrspacecast (generic->AS1), then bitcast within AS1
    return (g1_v4i*)(__attribute__((address_space(1))) void*)(void*)p;
}
__device__ __forceinline__ l3_v4i* as_lds_v4i(void* p) {
    return (l3_v4i*)(__attribute__((address_space(3))) void*)p;
}
__device__ __forceinline__ void wait_asynccnt0() {
#if defined(__has_builtin) && __has_builtin(__builtin_amdgcn_s_wait_asynccnt)
    __builtin_amdgcn_s_wait_asynccnt(0);
#else
    asm volatile("s_wait_asynccnt 0" ::: "memory");
#endif
}
#endif

__device__ __forceinline__ __bf16 tobf(float f) {
    unsigned u = __builtin_bit_cast(unsigned, f);
    unsigned r = (u + 0x7fffu + ((u >> 16) & 1u)) >> 16;   // round-to-nearest-even
    unsigned short s = (unsigned short)r;
    return __builtin_bit_cast(__bf16, s);
}

__device__ __forceinline__ v8f wmma_bf(const V16& a, const V16& b, v8f c) {
    // (neg_a, A, neg_b, B, c_mod, C, reuse_a, reuse_b)
    return __builtin_amdgcn_wmma_f32_16x16x32_bf16(false, a.v, false, b.v,
                                                   (short)0, c, false, false);
}

__device__ __forceinline__ v8f zero8() {
    v8f z = {0.f,0.f,0.f,0.f,0.f,0.f,0.f,0.f};
    return z;
}

// ---------------------------------------------------------------------------
// Kernel 1: f32 -> bf16 conversion (4 elems/thread)
// ---------------------------------------------------------------------------
__global__ void cvt_bf16_kernel(const float* __restrict__ in,
                                __bf16* __restrict__ out, int n) {
    int i = (blockIdx.x * blockDim.x + threadIdx.x) * 4;
    if (i < n) {
        float4 f = *(const float4*)(in + i);
        union { __bf16 e[4]; uint2 u; } p;
        p.e[0] = tobf(f.x); p.e[1] = tobf(f.y);
        p.e[2] = tobf(f.z); p.e[3] = tobf(f.w);
        *(uint2*)(out + i) = p.u;
    }
}

// ---------------------------------------------------------------------------
// Kernel 2/4: tiled bf16 WMMA GEMM.  C[M,N] = A[M,1024] * W[1024,N] + bias.
//   block tile 128x128, BK=32, 256 threads = 8 waves (4 M-waves x 2 N-waves),
//   each wave -> 32x64 output (2x4 grid of 16x16 f32 accumulators, 8 WMMA/step)
//   A tile staged via GLOBAL_LOAD_ASYNC_TO_LDS_B128 when available.
//   MODE 0: f32 output to outF (projection)
//   MODE 1: scatter bf16 Q [BH][T][D], K [BH][T][D], Vt [BH][D][T]
// ---------------------------------------------------------------------------
#define BM 128
#define BN 128
#define BK 32
#define LDA 40
#define LDB 40

template <int MODE>
__global__ __launch_bounds__(256)
void gemm_bf16_kernel(const __bf16* __restrict__ A, const __bf16* __restrict__ W,
                      const float* __restrict__ bias, int N,
                      float* __restrict__ outF,
                      __bf16* __restrict__ Qb, __bf16* __restrict__ Kb,
                      __bf16* __restrict__ Vt)
{
    __shared__ __bf16 As[BM * LDA];
    __shared__ __bf16 Bs[BN * LDB];

    const int tid  = threadIdx.x;
    const int lane = tid & 31;
    const int w    = tid >> 5;
    const int wm   = w & 3;       // 0..3 : 32-row slab
    const int wn   = w >> 2;      // 0..1 : 64-col slab
    const int half = lane >> 4;
    const int l15  = lane & 15;
    const int bn0  = blockIdx.x * BN;
    const int bm0  = blockIdx.y * BM;

    // cooperative-load mappings
    const int arow = tid >> 1, ac16 = (tid & 1) * 16;  // A: 128 rows x 32 halfs
    const int bkr  = tid >> 3, bn16 = (tid & 7) * 16;  // W:  32 rows x 128 halfs

    v8f acc[2][4];
#pragma unroll
    for (int i = 0; i < 2; ++i)
#pragma unroll
        for (int j = 0; j < 4; ++j) acc[i][j] = zero8();

    for (int kt = 0; kt < 1024; kt += BK) {
        const __bf16* ag = A + (size_t)(bm0 + arow) * 1024 + kt + ac16;
        const __bf16* wg = W + (size_t)(kt + bkr) * N + bn0 + bn16;
        uint4 bv0 = *(const uint4*)(wg);
        uint4 bv1 = *(const uint4*)(wg + 8);
        if (kt + BK < 1024) {                       // global_prefetch_b8 next tiles
            __builtin_prefetch(ag + BK, 0, 3);
            __builtin_prefetch(wg + (size_t)BK * N, 0, 3);
        }

#if USE_ASYNC_LDS
        __syncthreads();
        // A tile: memory -> LDS directly, no VGPR staging (ASYNCcnt-tracked)
        __builtin_amdgcn_global_load_async_to_lds_b128(
            as_global_v4i(ag),     as_lds_v4i(&As[arow * LDA + ac16]),     0, 0);
        __builtin_amdgcn_global_load_async_to_lds_b128(
            as_global_v4i(ag + 8), as_lds_v4i(&As[arow * LDA + ac16 + 8]), 0, 0);
        union { uint4 u[2]; __bf16 e[16]; } bb;
        bb.u[0] = bv0; bb.u[1] = bv1;
#pragma unroll
        for (int i = 0; i < 16; ++i)                // transpose into [n][k]
            Bs[(bn16 + i) * LDB + bkr] = bb.e[i];
        wait_asynccnt0();
        __syncthreads();
#else
        uint4 av0 = *(const uint4*)(ag);
        uint4 av1 = *(const uint4*)(ag + 8);
        __syncthreads();
        *(uint4*)&As[arow * LDA + ac16]     = av0;
        *(uint4*)&As[arow * LDA + ac16 + 8] = av1;
        union { uint4 u[2]; __bf16 e[16]; } bb;
        bb.u[0] = bv0; bb.u[1] = bv1;
#pragma unroll
        for (int i = 0; i < 16; ++i)                // transpose into [n][k]
            Bs[(bn16 + i) * LDB + bkr] = bb.e[i];
        __syncthreads();
#endif

        // A fragments: 16x32, lane row, K at half*8 and half*8+16
        V16 afr[2];
#pragma unroll
        for (int i = 0; i < 2; ++i) {
            const int r = wm * 32 + i * 16 + l15;
            afr[i].h[0] = *(const v8bf*)&As[r * LDA + half * 8];
            afr[i].h[1] = *(const v8bf*)&As[r * LDA + half * 8 + 16];
        }
#pragma unroll
        for (int j = 0; j < 4; ++j) {
            // B fragment: 32x16, lane col = n, K at half*8 and half*8+16
            const int n = wn * 64 + j * 16 + l15;
            V16 bfr;
            bfr.h[0] = *(const v8bf*)&Bs[n * LDB + half * 8];
            bfr.h[1] = *(const v8bf*)&Bs[n * LDB + half * 8 + 16];
#pragma unroll
            for (int i = 0; i < 2; ++i)
                acc[i][j] = wmma_bf(afr[i], bfr, acc[i][j]);
        }
    }

    // epilogue.  C/D layout: VGPR r -> rows r (lanes 0-15), r+8 (lanes 16-31),
    // N = lane&15.
#pragma unroll
    for (int j = 0; j < 4; ++j) {
        const int colb = bn0 + wn * 64 + j * 16;          // wave-uniform
        const int col  = colb + l15;
        const float bv = bias[col];
        // wave-uniform section select -> scalar branch
        const int sec = __builtin_amdgcn_readfirstlane(colb) >> 10;
#pragma unroll
        for (int i = 0; i < 2; ++i) {
#pragma unroll
            for (int r = 0; r < 8; ++r) {
                const int row = bm0 + wm * 32 + i * 16 + r + half * 8;
                const float v = acc[i][j][r] + bv;
                if (MODE == 0) {
                    outF[(size_t)row * N + col] = v;
                } else {
                    const int cc  = col & 1023;
                    const int hh  = cc >> 6, dd = cc & 63;
                    const int bat = row >> 11, tt = row & 2047;
                    const int bh  = bat * 16 + hh;
                    const __bf16 hv = tobf(v);
                    if (sec == 0)      Qb[((size_t)bh * 2048 + tt) * 64 + dd] = hv;
                    else if (sec == 1) Kb[((size_t)bh * 2048 + tt) * 64 + dd] = hv;
                    else               Vt[((size_t)bh * 64 + dd) * 2048 + tt] = hv;
                }
            }
        }
    }
}

// ---------------------------------------------------------------------------
// Kernel 3: flash attention.  grid = (BH=32, T/128=16), 128 threads = 4 waves.
// Wave w owns 32 query rows [qr0, qr0+31] (two 16-row tiles).  Online softmax
// over 32-key blocks; K/V fragments shared by both row tiles (16 WMMA/block).
// ---------------------------------------------------------------------------
__global__ __launch_bounds__(128)
void flash_attn_kernel(const __bf16* __restrict__ Q, const __bf16* __restrict__ K,
                       const __bf16* __restrict__ Vt, __bf16* __restrict__ O)
{
    __shared__ __bf16 Pb[4][2][16 * 40];  // per-wave, per-row-tile P staging

    const int tid  = threadIdx.x;
    const int lane = tid & 31;
    const int w    = tid >> 5;
    const int half = lane >> 4;
    const int l15  = lane & 15;
    const int bh   = blockIdx.x;            // 0..31
    const int bat  = bh >> 4, h = bh & 15;
    const int qr0  = blockIdx.y * 128 + w * 32;

    const __bf16* Qp = Q  + (size_t)bh * 2048 * 64;
    const __bf16* Kp = K  + (size_t)bh * 2048 * 64;
    const __bf16* Vp = Vt + (size_t)bh * 64 * 2048;

    // Q fragments: [row-tile i][d-half c]
    V16 qf[2][2];
#pragma unroll
    for (int i = 0; i < 2; ++i) {
        const int row = qr0 + i * 16 + l15;
#pragma unroll
        for (int c = 0; c < 2; ++c) {
            qf[i][c].h[0] = *(const v8bf*)(Qp + (size_t)row * 64 + c * 32 + half * 8);
            qf[i][c].h[1] = *(const v8bf*)(Qp + (size_t)row * 64 + c * 32 + half * 8 + 16);
        }
    }

    v8f o[2][4];
    float m[2][8], l[2][8];
#pragma unroll
    for (int i = 0; i < 2; ++i) {
#pragma unroll
        for (int j = 0; j < 4; ++j) o[i][j] = zero8();
#pragma unroll
        for (int r = 0; r < 8; ++r) { m[i][r] = -1e30f; l[i][r] = 0.f; }
    }

    const float scale = 0.03125f;            // 1/sqrt(1024)
    const int nkb = (qr0 + 63) >> 5;         // 32-key blocks covering keys <= qr0+31

    for (int it = 0; it < nkb; ++it) {
        const int kb = it * 32;

        // ---- S[i][s] = Q_i @ K_s^T, two 16-key subtiles -------------------
        v8f S[2][2];
#pragma unroll
        for (int s = 0; s < 2; ++s) {
            const int key = kb + s * 16 + l15;
            V16 kf0, kf1;
            kf0.h[0] = *(const v8bf*)(Kp + (size_t)key * 64 + 0  + half * 8);
            kf0.h[1] = *(const v8bf*)(Kp + (size_t)key * 64 + 0  + half * 8 + 16);
            kf1.h[0] = *(const v8bf*)(Kp + (size_t)key * 64 + 32 + half * 8);
            kf1.h[1] = *(const v8bf*)(Kp + (size_t)key * 64 + 32 + half * 8 + 16);
#pragma unroll
            for (int i = 0; i < 2; ++i) {
                S[i][s] = wmma_bf(qf[i][0], kf0, zero8());
                S[i][s] = wmma_bf(qf[i][1], kf1, S[i][s]);
            }
        }

        // ---- online softmax update (per row tile, per row r / r+8) -------
#pragma unroll
        for (int i = 0; i < 2; ++i) {
            __bf16* Pw = &Pb[w][i][0];
#pragma unroll
            for (int r = 0; r < 8; ++r) {
                const int row = qr0 + i * 16 + r + half * 8;
                float s0 = S[i][0][r] * scale;
                float s1 = S[i][1][r] * scale;
                if (kb + l15 > row)      s0 = -1e30f;   // causal mask
                if (kb + 16 + l15 > row) s1 = -1e30f;
                float t = fmaxf(s0, s1);
                t = fmaxf(t, __shfl_xor(t, 1));
                t = fmaxf(t, __shfl_xor(t, 2));
                t = fmaxf(t, __shfl_xor(t, 4));
                t = fmaxf(t, __shfl_xor(t, 8));         // row max in 16-lane half
                const float mnew  = fmaxf(m[i][r], t);
                const float p0    = __expf(s0 - mnew);
                const float p1    = __expf(s1 - mnew);
                const float alpha = __expf(m[i][r] - mnew);
                m[i][r] = mnew;
                float u = p0 + p1;
                u += __shfl_xor(u, 1);
                u += __shfl_xor(u, 2);
                u += __shfl_xor(u, 4);
                u += __shfl_xor(u, 8);                  // row sum
                l[i][r] = l[i][r] * alpha + u;
#pragma unroll
                for (int j = 0; j < 4; ++j) o[i][j][r] *= alpha;
                const int rl = r + half * 8;            // C-layout row -> LDS row
                Pw[rl * 40 + l15]      = tobf(p0);
                Pw[rl * 40 + 16 + l15] = tobf(p1);
            }
        }
        asm volatile("s_wait_dscnt 0" ::: "memory");    // LDS RAW within wave

        // ---- O += P @ V (V fragments shared by both row tiles) -----------
        V16 pf[2];
#pragma unroll
        for (int i = 0; i < 2; ++i) {
            const __bf16* Pw = &Pb[w][i][0];
            pf[i].h[0] = *(const v8bf*)&Pw[l15 * 40 + half * 8];
            pf[i].h[1] = *(const v8bf*)&Pw[l15 * 40 + half * 8 + 16];
        }
#pragma unroll
        for (int j = 0; j < 4; ++j) {
            const int d = j * 16 + l15;
            V16 vf;
            vf.h[0] = *(const v8bf*)(Vp + (size_t)d * 2048 + kb + half * 8);
            vf.h[1] = *(const v8bf*)(Vp + (size_t)d * 2048 + kb + half * 8 + 16);
#pragma unroll
            for (int i = 0; i < 2; ++i)
                o[i][j] = wmma_bf(pf[i], vf, o[i][j]);
        }
    }

    // ---- normalize + store bf16 to [B,T,C] -------------------------------
#pragma unroll
    for (int i = 0; i < 2; ++i) {
#pragma unroll
        for (int r = 0; r < 8; ++r) {
            const float inv = 1.0f / l[i][r];
            const int trow  = qr0 + i * 16 + r + half * 8;
#pragma unroll
            for (int j = 0; j < 4; ++j) {
                O[((size_t)(bat * 2048 + trow)) * 1024 + h * 64 + j * 16 + l15] =
                    tobf(o[i][j][r] * inv);
            }
        }
    }
}

// ---------------------------------------------------------------------------
// Host-side launcher
// ---------------------------------------------------------------------------
extern "C" void kernel_launch(void* const* d_in, const int* in_sizes, int n_in,
                              void* d_out, int out_size, void* d_ws, size_t ws_size,
                              hipStream_t stream) {
    const float* x      = (const float*)d_in[0];   // [2,2048,1024]
    const float* W_attn = (const float*)d_in[1];   // [1024,3072]
    const float* b_attn = (const float*)d_in[2];   // [3072]
    const float* W_proj = (const float*)d_in[3];   // [1024,1024]
    const float* b_proj = (const float*)d_in[4];   // [1024]
    float* out = (float*)d_out;                    // [2,2048,1024] f32

    char* ws = (char*)d_ws;
    __bf16* xb   = (__bf16*)ws; ws += (size_t)4096 * 1024 * 2;
    __bf16* wab  = (__bf16*)ws; ws += (size_t)1024 * 3072 * 2;
    __bf16* wpb  = (__bf16*)ws; ws += (size_t)1024 * 1024 * 2;
    __bf16* Qb   = (__bf16*)ws; ws += (size_t)32 * 2048 * 64 * 2;
    __bf16* Kb   = (__bf16*)ws; ws += (size_t)32 * 2048 * 64 * 2;
    __bf16* Vt   = (__bf16*)ws; ws += (size_t)32 * 64 * 2048 * 2;
    __bf16* Ab   = (__bf16*)ws; ws += (size_t)4096 * 1024 * 2;   // pre-proj attn out

    // 1) convert inputs to bf16
    cvt_bf16_kernel<<<(4096 * 1024) / 1024, 256, 0, stream>>>(x, xb, 4096 * 1024);
    cvt_bf16_kernel<<<(1024 * 3072) / 1024, 256, 0, stream>>>(W_attn, wab, 1024 * 3072);
    cvt_bf16_kernel<<<(1024 * 1024) / 1024, 256, 0, stream>>>(W_proj, wpb, 1024 * 1024);

    // 2) QKV GEMM -> Q, K, V^T  (M=4096, K=1024, N=3072)
    gemm_bf16_kernel<1><<<dim3(3072 / BN, 4096 / BM), 256, 0, stream>>>(
        xb, wab, b_attn, 3072, nullptr, Qb, Kb, Vt);

    // 3) flash attention -> Ab [4096,1024] bf16
    flash_attn_kernel<<<dim3(32, 16), 128, 0, stream>>>(Qb, Kb, Vt, Ab);

    // 4) projection GEMM -> out f32  (M=4096, K=1024, N=1024)
    gemm_bf16_kernel<0><<<dim3(1024 / BN, 4096 / BM), 256, 0, stream>>>(
        Ab, wpb, b_proj, 1024, out, nullptr, nullptr, nullptr);
}
